// GIN_15719580303914
// MI455X (gfx1250) — matrix operand order repaired and checked by reading, not compiled, not run on hardware
//
#include <hip/hip_runtime.h>
#include <math.h>

// CDNA5 / gfx1250. wave32. WMMA f16 -> f32 accumulate.
typedef __attribute__((ext_vector_type(16))) _Float16 v16h;
typedef __attribute__((ext_vector_type(8)))  _Float16 v8h;
typedef __attribute__((ext_vector_type(8)))  float    v8f;

#define FEAT 128
#define NCLS 10
#define BN_EPS 1e-5f

// ---------------------------------------------------------------------------
// agg = h  (vectorized f32 copy, grid-stride)
// ---------------------------------------------------------------------------
__global__ __launch_bounds__(256)
void copy_f32(const float* __restrict__ src, float* __restrict__ dst, long n4) {
  long i      = (long)blockIdx.x * blockDim.x + threadIdx.x;
  long stride = (long)gridDim.x * blockDim.x;
  const float4* s4 = (const float4*)src;
  float4*       d4 = (float4*)dst;
  for (; i < n4; i += stride) d4[i] = s4[i];
}

// ---------------------------------------------------------------------------
// agg[dst] += h[src] over edges. 2 edges x 128 feats per 256-thread block.
// Coalesced gather + coalesced native f32 atomics (global_atomic_add_f32).
// ---------------------------------------------------------------------------
__global__ __launch_bounds__(256)
void scatter_add(const float* __restrict__ h, const int* __restrict__ edges,
                 float* __restrict__ agg, int nE) {
  int e = blockIdx.x * 2 + (threadIdx.x >> 7);
  int f = threadIdx.x & 127;
  if (e < nE) {
    int s = edges[e];        // src row of edge_index
    int d = edges[nE + e];   // dst row of edge_index
    unsafeAtomicAdd(&agg[(size_t)d * FEAT + f], h[(size_t)s * FEAT + f]);
  }
}

// ---------------------------------------------------------------------------
// OUT = epilogue(IN @ W^T + bias), IN:[N,128] (f32 or f16), W:[128,128] f32.
// 128 threads = 4 waves; each wave -> 16 rows x 128 cols via 8 WMMA tiles,
// K in 4 steps of 32. W staged in LDS as f16 (32 KB).
//   DO_BN=0: out f16 = relu(z)
//   DO_BN=1: out f32 = gamma*(relu(z)-rmean)*rsqrt(rvar+eps)+beta
// ---------------------------------------------------------------------------
template <bool IN_HALF, bool DO_BN>
__global__ __launch_bounds__(128)
void gemm128(const void* __restrict__ in, const float* __restrict__ W,
             const float* __restrict__ bias,
             const float* __restrict__ gamma, const float* __restrict__ beta,
             const float* __restrict__ rmean, const float* __restrict__ rvar,
             void* __restrict__ out, int N) {
  __shared__ _Float16 ldsW[FEAT * FEAT];

  // Stage W (f32 -> f16) into LDS, coalesced float4 reads.
  const float4* W4 = (const float4*)W;
  for (int i = threadIdx.x; i < FEAT * FEAT / 4; i += 128) {
    float4 w = W4[i];
    _Float16* p = &ldsW[i * 4];
    p[0] = (_Float16)w.x; p[1] = (_Float16)w.y;
    p[2] = (_Float16)w.z; p[3] = (_Float16)w.w;
  }
  __syncthreads();

  const int lane    = threadIdx.x & 31;
  const int wave    = threadIdx.x >> 5;
  const int rowBase = blockIdx.x * 64 + wave * 16;
  const int m  = lane & 15;   // A row-in-tile / B,C column-in-tile
  const int hh = lane >> 4;   // K-half selector (A,B); row-half selector (C)

  int arow = rowBase + m;     // clamp OOB rows for loads; stores are guarded
  if (arow > N - 1) arow = N - 1;

  // Accumulators: init with bias broadcast (each lane's fragment is one column).
  v8f c[8];
#pragma unroll
  for (int nt = 0; nt < 8; ++nt) {
    float bv = bias[nt * 16 + m];
#pragma unroll
    for (int r = 0; r < 8; ++r) c[nt][r] = bv;
  }

#pragma unroll
  for (int kb = 0; kb < FEAT; kb += 32) {
    // A fragment: lane holds K = {hh*8..hh*8+7} U {16+hh*8..16+hh*8+7}
    v16h a;
    if (IN_HALF) {
      const _Float16* ap = (const _Float16*)in + (size_t)arow * FEAT + kb;
      v8h x0 = *(const v8h*)(ap + hh * 8);
      v8h x1 = *(const v8h*)(ap + 16 + hh * 8);
#pragma unroll
      for (int i = 0; i < 8; ++i) { a[i] = x0[i]; a[8 + i] = x1[i]; }
    } else {
      const float* ap = (const float*)in + (size_t)arow * FEAT + kb;
      float4 x0 = *(const float4*)(ap + hh * 8);
      float4 x1 = *(const float4*)(ap + hh * 8 + 4);
      float4 x2 = *(const float4*)(ap + 16 + hh * 8);
      float4 x3 = *(const float4*)(ap + 16 + hh * 8 + 4);
      a[0]=(_Float16)x0.x; a[1]=(_Float16)x0.y; a[2]=(_Float16)x0.z; a[3]=(_Float16)x0.w;
      a[4]=(_Float16)x1.x; a[5]=(_Float16)x1.y; a[6]=(_Float16)x1.z; a[7]=(_Float16)x1.w;
      a[8]=(_Float16)x2.x; a[9]=(_Float16)x2.y; a[10]=(_Float16)x2.z; a[11]=(_Float16)x2.w;
      a[12]=(_Float16)x3.x; a[13]=(_Float16)x3.y; a[14]=(_Float16)x3.z; a[15]=(_Float16)x3.w;
    }

#pragma unroll
    for (int nt = 0; nt < 8; ++nt) {
      // B fragment: column n = nt*16+m, K = {hh*8..hh*8+7} U {16+hh*8..16+hh*8+7}
      const _Float16* bp = &ldsW[(nt * 16 + m) * FEAT + kb];
      v8h y0 = *(const v8h*)(bp + hh * 8);
      v8h y1 = *(const v8h*)(bp + 16 + hh * 8);
      v16h b;
#pragma unroll
      for (int i = 0; i < 8; ++i) { b[i] = y0[i]; b[8 + i] = y1[i]; }
      c[nt] = __builtin_amdgcn_wmma_f32_16x16x32_f16(
          /*neg_a=*/false, a, /*neg_b=*/false, b,
          /*c_mod=*/(short)0, c[nt], /*reuse_a=*/false, /*reuse_b=*/false);
    }
  }

  // Epilogue: relu (+BN). C/D layout: lane -> col n=m, elem r -> row hh*8+r.
#pragma unroll
  for (int nt = 0; nt < 8; ++nt) {
    const int col = nt * 16 + m;
    float sc = 1.f, sh = 0.f;
    if (DO_BN) {
      sc = gamma[col] * rsqrtf(rvar[col] + BN_EPS);
      sh = beta[col] - rmean[col] * sc;
    }
#pragma unroll
    for (int r = 0; r < 8; ++r) {
      int row = rowBase + hh * 8 + r;
      if (row < N) {
        float z = c[nt][r];
        z = z > 0.f ? z : 0.f;
        if (DO_BN)
          ((float*)out)[(size_t)row * FEAT + col] = z * sc + sh;
        else
          ((_Float16*)out)[(size_t)row * FEAT + col] = (_Float16)z;
      }
    }
  }
}

// ---------------------------------------------------------------------------
// logits = t @ lin2_w^T + lin2_b (128 -> 10), then log_softmax. lin2_w in LDS.
// ---------------------------------------------------------------------------
__global__ __launch_bounds__(256)
void head_logits(const _Float16* __restrict__ t, const float* __restrict__ w,
                 const float* __restrict__ b, float* __restrict__ out, int N) {
  __shared__ float lw[NCLS * FEAT];
  __shared__ float lb[NCLS];
  for (int i = threadIdx.x; i < NCLS * FEAT; i += 256) lw[i] = w[i];
  if (threadIdx.x < NCLS) lb[threadIdx.x] = b[threadIdx.x];
  __syncthreads();

  int node   = blockIdx.x * 256 + threadIdx.x;
  int stride = gridDim.x * 256;
  for (; node < N; node += stride) {
    float acc[NCLS];
#pragma unroll
    for (int ci = 0; ci < NCLS; ++ci) acc[ci] = lb[ci];
    const _Float16* xp = t + (size_t)node * FEAT;
    for (int k = 0; k < FEAT; ++k) {
      float xv = (float)xp[k];
#pragma unroll
      for (int ci = 0; ci < NCLS; ++ci) acc[ci] += xv * lw[ci * FEAT + k];
    }
    float mx = acc[0];
#pragma unroll
    for (int ci = 1; ci < NCLS; ++ci) mx = fmaxf(mx, acc[ci]);
    float s = 0.f;
#pragma unroll
    for (int ci = 0; ci < NCLS; ++ci) s += __expf(acc[ci] - mx);
    float lse = mx + __logf(s);
#pragma unroll
    for (int ci = 0; ci < NCLS; ++ci)
      out[(size_t)node * NCLS + ci] = acc[ci] - lse;
  }
}

// ---------------------------------------------------------------------------
extern "C" void kernel_launch(void* const* d_in, const int* in_sizes, int n_in,
                              void* d_out, int out_size, void* d_ws, size_t ws_size,
                              hipStream_t stream) {
  const float* x     = (const float*)d_in[0];
  const int*   edges = (const int*)d_in[1];   // [2, nE] flat, int32
  const float* W1    = (const float*)d_in[2];
  const float* b1    = (const float*)d_in[3];
  const float* W2    = (const float*)d_in[4];
  const float* b2    = (const float*)d_in[5];
  const float* gamma = (const float*)d_in[6];
  const float* beta  = (const float*)d_in[7];
  const float* rmean = (const float*)d_in[8];
  const float* rvar  = (const float*)d_in[9];
  const float* lin1w = (const float*)d_in[10];
  const float* lin1b = (const float*)d_in[11];
  const float* lin2w = (const float*)d_in[12];
  const float* lin2b = (const float*)d_in[13];

  const int N  = in_sizes[0] / FEAT;
  const int nE = in_sizes[1] / 2;

  // Workspace layout: agg f32 | hbuf f32 | z1 f16  (needs N*128*10 bytes)
  char* ws = (char*)d_ws;
  float*     agg  = (float*)ws;
  float*     hbuf = (float*)(ws + (size_t)N * FEAT * 4);
  _Float16*  z1   = (_Float16*)(ws + (size_t)N * FEAT * 8);

  const int  gemmGrid    = (N + 63) / 64;
  const int  scatterGrid = (nE + 1) / 2;
  const long n4          = (long)N * FEAT / 4;

  const float* hcur = x;
  for (int L = 0; L < 3; ++L) {
    // z = h + segment_sum(h[src] -> dst)
    copy_f32<<<2048, 256, 0, stream>>>(hcur, agg, n4);
    scatter_add<<<scatterGrid, 256, 0, stream>>>(hcur, edges, agg, nE);
    // relu(z @ W1^T + b1) -> z1 (f16)
    gemm128<false, false><<<gemmGrid, 128, 0, stream>>>(
        agg, W1 + (size_t)L * FEAT * FEAT, b1 + L * FEAT,
        nullptr, nullptr, nullptr, nullptr, z1, N);
    // BN(relu(z1 @ W2^T + b2)) -> hbuf (f32)
    gemm128<true, true><<<gemmGrid, 128, 0, stream>>>(
        z1, W2 + (size_t)L * FEAT * FEAT, b2 + L * FEAT,
        gamma + L * FEAT, beta + L * FEAT, rmean + L * FEAT, rvar + L * FEAT,
        hbuf, N);
    hcur = hbuf;
  }

  // relu(h @ lin1_w^T + lin1_b) -> z1 (f16)
  gemm128<false, false><<<gemmGrid, 128, 0, stream>>>(
      hcur, lin1w, lin1b, nullptr, nullptr, nullptr, nullptr, z1, N);
  // log_softmax(z1 @ lin2_w^T + lin2_b) -> out
  head_logits<<<512, 256, 0, stream>>>(z1, lin2w, lin2b, (float*)d_out, N);
}
